// TopKMoE_56710748176710
// MI455X (gfx1250) — matrix operand order repaired and checked by reading, not compiled, not run on hardware
//
#include <hip/hip_runtime.h>

// ---------------------------------------------------------------------------
// Problem constants (from reference): B=8, T=8192, D=512, H=512, N=32, K=2
// ---------------------------------------------------------------------------
#define BT      65536      // B*T tokens
#define BTK     131072     // BT * TOP_K assignments
#define DDIM    512
#define HDIM    512
#define NEXP    32
#define CAP     5120       // ceil(1.25 * BTK / N)
#define NC      (NEXP * CAP)   // 163840 capacity rows

#define KDIM    512        // K of both expert GEMMs
#define KSTEPS  (KDIM / 32)
#define BCOL_STRIDE 40     // padded LDS column stride (elements); 80B = 20 banks

typedef __attribute__((ext_vector_type(16))) __bf16 v16bf;
typedef __attribute__((ext_vector_type(8)))  __bf16 v8bf;
typedef __attribute__((ext_vector_type(8)))  float  v8f;
typedef __attribute__((ext_vector_type(4)))  unsigned short us4;

__device__ __forceinline__ unsigned short f2bf(float f) {
    unsigned int u = __float_as_uint(f);
    unsigned int r = u + 0x7FFFu + ((u >> 16) & 1u);   // round-to-nearest-even
    return (unsigned short)(r >> 16);
}
__device__ __forceinline__ float bf2f(unsigned short h) {
    return __uint_as_float(((unsigned int)h) << 16);
}

// ---------------------------------------------------------------------------
// Kernel: 32x32 tiled transpose + f32 -> bf16  (per-matrix via blockIdx.z)
// ---------------------------------------------------------------------------
__global__ __launch_bounds__(256)
void k_transpose_cvt(const float* __restrict__ src, unsigned short* __restrict__ dst,
                     int R, int Cc) {
    __shared__ float tile[32][33];
    const size_t mat = (size_t)blockIdx.z * (size_t)R * (size_t)Cc;
    src += mat; dst += mat;
    const int tx = threadIdx.x;        // 0..31
    const int ty = threadIdx.y;        // 0..7
    const int c0 = blockIdx.x * 32;
    const int r0 = blockIdx.y * 32;
#pragma unroll
    for (int k = 0; k < 4; ++k)
        tile[ty + 8 * k][tx] = src[(size_t)(r0 + ty + 8 * k) * Cc + (c0 + tx)];
    __syncthreads();
#pragma unroll
    for (int k = 0; k < 4; ++k)
        dst[(size_t)(c0 + ty + 8 * k) * R + (r0 + tx)] = f2bf(tile[tx][ty + 8 * k]);
}

// ---------------------------------------------------------------------------
// Kernel: gating.  One wave32 per token; lane = expert.
// ---------------------------------------------------------------------------
__global__ __launch_bounds__(256)
void k_gate(const float* __restrict__ x, const float* __restrict__ Wg,
            int* __restrict__ expert_id, float* __restrict__ gate_w) {
    const int wave = threadIdx.x >> 5;
    const int lane = threadIdx.x & 31;
    const int t = blockIdx.x * 8 + wave;
    if (t >= BT) return;

    const float* xr = x + (size_t)t * DDIM;
    float acc = 0.f;
#pragma unroll 4
    for (int d = 0; d < DDIM; d += 4) {
        float4 xv = *(const float4*)(xr + d);
        acc += xv.x * Wg[(d + 0) * NEXP + lane];
        acc += xv.y * Wg[(d + 1) * NEXP + lane];
        acc += xv.z * Wg[(d + 2) * NEXP + lane];
        acc += xv.w * Wg[(d + 3) * NEXP + lane];
    }
    float m = acc;
#pragma unroll
    for (int off = 16; off > 0; off >>= 1) m = fmaxf(m, __shfl_xor(m, off, 32));
    float p = __expf(acc - m);
    float s = p;
#pragma unroll
    for (int off = 16; off > 0; off >>= 1) s += __shfl_xor(s, off, 32);
    const float prob = p / s;

    float v1 = prob; int a1 = lane;
#pragma unroll
    for (int off = 16; off > 0; off >>= 1) {
        float ov = __shfl_xor(v1, off, 32);
        int   oa = __shfl_xor(a1, off, 32);
        if (ov > v1 || (ov == v1 && oa < a1)) { v1 = ov; a1 = oa; }
    }
    float v2 = (lane == a1) ? -3.0e38f : prob; int a2 = lane;
#pragma unroll
    for (int off = 16; off > 0; off >>= 1) {
        float ov = __shfl_xor(v2, off, 32);
        int   oa = __shfl_xor(a2, off, 32);
        if (ov > v2 || (ov == v2 && oa < a2)) { v2 = ov; a2 = oa; }
    }
    if (lane == 0) {
        float denom = fmaxf(v1 + v2, 1e-9f);
        expert_id[2 * t + 0] = a1;
        expert_id[2 * t + 1] = a2;
        gate_w[2 * t + 0] = v1 / denom;
        gate_w[2 * t + 1] = v2 / denom;
    }
}

// ---------------------------------------------------------------------------
// Kernel: order-preserving rank within expert (block prefix-sum scan).
// ---------------------------------------------------------------------------
__global__ __launch_bounds__(256)
void k_rank(const int* __restrict__ expert_id, int* __restrict__ dest,
            int* __restrict__ slot_token) {
    __shared__ int sscan[256];
    const int e = blockIdx.x;
    const int tid = threadIdx.x;
    int count = 0;
    for (int base = 0; base < BTK; base += 256) {
        const int i = base + tid;
        const int flag = (expert_id[i] == e) ? 1 : 0;
        sscan[tid] = flag;
        __syncthreads();
#pragma unroll
        for (int off = 1; off < 256; off <<= 1) {
            int v = (tid >= off) ? sscan[tid - off] : 0;
            __syncthreads();
            sscan[tid] += v;
            __syncthreads();
        }
        const int incl = sscan[tid];
        const int total = sscan[255];
        if (flag) {
            const int pos = count + incl - 1;
            int d = -1;
            if (pos < CAP) {
                d = e * CAP + pos;
                slot_token[d] = i >> 1;
            }
            dest[i] = d;
        }
        count += total;
        __syncthreads();
    }
    for (int c = count + tid; c < CAP; c += 256) slot_token[e * CAP + c] = -1;
}

// ---------------------------------------------------------------------------
// Kernel: dispatch tokens into bf16 capacity buffer [NC, D] (zeros if empty)
// ---------------------------------------------------------------------------
__global__ __launch_bounds__(128)
void k_dispatch(const float* __restrict__ x, const int* __restrict__ slot_token,
                unsigned short* __restrict__ bufA) {
    const int row = blockIdx.x;
    const int tok = slot_token[row];
    const int d = threadIdx.x * 4;
    us4 o;
    if (tok >= 0) {
        float4 xv = *(const float4*)(x + (size_t)tok * DDIM + d);
        o.x = f2bf(xv.x); o.y = f2bf(xv.y); o.z = f2bf(xv.z); o.w = f2bf(xv.w);
    } else {
        o.x = 0; o.y = 0; o.z = 0; o.w = 0;
    }
    *(us4*)(bufA + (size_t)row * DDIM + d) = o;
}

// ---------------------------------------------------------------------------
// One k-step of the WMMA GEMM pipeline.
//   bsb:      LDS buffer holding the CURRENT 64x32 B tile
//   ldst:     this thread's 16B slot in the NEXT LDS buffer
//   aCur:     A fragment for this step (consumed)
//   aNext:    A fragment register for the prefetched step (produced)
//   PREFETCH: compile-time: issue global loads for step kkNext + LDS store
// ---------------------------------------------------------------------------
template<bool PREFETCH>
__device__ __forceinline__ void gemm_step(
    const unsigned short* __restrict__ bsb, unsigned short* __restrict__ ldst,
    const unsigned short* __restrict__ arow, const unsigned short* __restrict__ sgp,
    int kkNext, int l15, int h, v16bf& aCur, v16bf& aNext, v8f acc[4]) {
    v8bf bNext;
    if (PREFETCH) {
        bNext = *(const v8bf*)(sgp + kkNext);
        ((v8bf*)&aNext)[0] = *(const v8bf*)(arow + kkNext + 8 * h);
        ((v8bf*)&aNext)[1] = *(const v8bf*)(arow + kkNext + 16 + 8 * h);
    }
    // fetch all four B fragments from LDS ...
    v16bf bfrag[4];
#pragma unroll
    for (int f = 0; f < 4; ++f) {
        const int off = (f * 16 + l15) * BCOL_STRIDE + 8 * h;
        ((v8bf*)&bfrag[f])[0] = *(const v8bf*)(bsb + off);
        ((v8bf*)&bfrag[f])[1] = *(const v8bf*)(bsb + off + 16);
    }
    // ... then the WMMA chain (distinct accumulators -> no D->A/B hazard)
#pragma unroll
    for (int f = 0; f < 4; ++f)
        acc[f] = __builtin_amdgcn_wmma_f32_16x16x32_bf16(
            false, aCur, false, bfrag[f], (short)0, acc[f], false, false);
    if (PREFETCH)
        *(v8bf*)ldst = bNext;                       // loadcnt wait lands here
#if __has_builtin(__builtin_amdgcn_sched_group_barrier)
    if (PREFETCH)
        __builtin_amdgcn_sched_group_barrier(0x020, 3, 0);  // 3 global reads first
    __builtin_amdgcn_sched_group_barrier(0x100, 8, 0);      // 8 LDS reads
    __builtin_amdgcn_sched_group_barrier(0x008, 4, 0);      // 4 WMMAs back-to-back
    if (PREFETCH)
        __builtin_amdgcn_sched_group_barrier(0x200, 1, 0);  // LDS write last
#endif
    __syncthreads();    // one barrier/step: read(i) vs write(i+2) separated by 2
}

// ---------------------------------------------------------------------------
// Kernel: grouped NT GEMM with v_wmma_f32_16x16x32_bf16.  K = 512 fixed.
//   Per expert n (blockIdx.z):  Out[m,c] = act( sum_k A[m,k]*Bt[c,k] + bias[c] )
//   Block = 8 waves -> 128(M) x 64(N) tile; wave = 16x64 strip (4 accumulators)
//   B tile in double-buffered LDS; fully unrolled manual ping-pong so buffer
//   indices / prefetch offsets are immediates and A never needs reg copies.
// ---------------------------------------------------------------------------
template<int DO_RELU>
__global__ __launch_bounds__(256)
void k_gemm_bf16(const unsigned short* __restrict__ A,
                 const unsigned short* __restrict__ Bt,
                 const float* __restrict__ bias,
                 unsigned short* __restrict__ Out,
                 int Ncols, int Mper) {
    __shared__ __align__(16) unsigned short Bs[2][64 * BCOL_STRIDE];

    const int n = blockIdx.z;
    const unsigned short* Ab = A  + (size_t)n * Mper * KDIM;
    const unsigned short* Bb = Bt + (size_t)n * Ncols * KDIM;
    const float* biasb = bias + (size_t)n * Ncols;
    unsigned short* Ob = Out + (size_t)n * Mper * Ncols;

    const int tid  = threadIdx.x;
    const int wave = tid >> 5;
    const int lane = tid & 31;
    const int l15  = lane & 15;
    const int h    = lane >> 4;
    const int m0   = blockIdx.x * 128 + wave * 16;
    const int nc0  = blockIdx.y * 64;

    // B staging: thread -> one 16B chunk of the 64x32 tile per k-step
    const int scol  = tid >> 2;                 // 0..63
    const int spart = tid & 3;                  // 0..3
    const unsigned short* sgp = Bb + (size_t)(nc0 + scol) * KDIM + spart * 8;
    const int sloff = scol * BCOL_STRIDE + spart * 8;

    const unsigned short* arow = Ab + (size_t)(m0 + l15) * KDIM;

    // prologue: stage k-step 0
    v8bf b0 = *(const v8bf*)(sgp);
    v16bf aCur, aNext;
    ((v8bf*)&aCur)[0] = *(const v8bf*)(arow + 8 * h);
    ((v8bf*)&aCur)[1] = *(const v8bf*)(arow + 16 + 8 * h);
    *(v8bf*)(&Bs[0][sloff]) = b0;
    __syncthreads();

    v8f acc[4] = {};
#pragma unroll
    for (int it = 0; it < KSTEPS; it += 2) {
        // even step: consume Bs[0]/aCur, prefetch (it+1) into Bs[1]/aNext
        gemm_step<true>(&Bs[0][0], &Bs[1][sloff], arow, sgp,
                        (it + 1) * 32, l15, h, aCur, aNext, acc);
        // odd step: consume Bs[1]/aNext, prefetch (it+2) into Bs[0]/aCur
        if (it + 2 < KSTEPS)
            gemm_step<true>(&Bs[1][0], &Bs[0][sloff], arow, sgp,
                            (it + 2) * 32, l15, h, aNext, aCur, acc);
        else
            gemm_step<false>(&Bs[1][0], &Bs[0][sloff], arow, sgp,
                             0, l15, h, aNext, aCur, acc);
    }

    // epilogue: D layout -> VGPR r holds row m0 + r + 8h, col nc0+f*16+l15
#pragma unroll
    for (int f = 0; f < 4; ++f) {
        const int col = nc0 + f * 16 + l15;
        const float bs = biasb[col];
#pragma unroll
        for (int r = 0; r < 8; ++r) {
            float v = acc[f][r] + bs;
            if (DO_RELU) v = fmaxf(v, 0.f);
            Ob[(size_t)(m0 + r + 8 * h) * Ncols + col] = f2bf(v);
        }
    }
}

// ---------------------------------------------------------------------------
// Kernel: combine.  y[t] = sum_k w_k * o[dest_k]  (passthrough if no survivor)
// ---------------------------------------------------------------------------
__global__ __launch_bounds__(128)
void k_combine(const float* __restrict__ x, const unsigned short* __restrict__ obuf,
               const int* __restrict__ dest, const float* __restrict__ gate_w,
               float* __restrict__ y) {
    const int t = blockIdx.x;
    const int d = threadIdx.x * 4;
    const int d0 = dest[2 * t + 0];
    const int d1 = dest[2 * t + 1];
    const float w0 = (d0 >= 0) ? gate_w[2 * t + 0] : 0.f;
    const float w1 = (d1 >= 0) ? gate_w[2 * t + 1] : 0.f;
    float4 out;
    if (w0 + w1 <= 1e-12f) {
        out = *(const float4*)(x + (size_t)t * DDIM + d);
    } else {
        const int r0 = (d0 >= 0) ? d0 : 0;
        const int r1 = (d1 >= 0) ? d1 : 0;
        us4 o0 = *(const us4*)(obuf + (size_t)r0 * DDIM + d);
        us4 o1 = *(const us4*)(obuf + (size_t)r1 * DDIM + d);
        out.x = w0 * bf2f(o0.x) + w1 * bf2f(o1.x);
        out.y = w0 * bf2f(o0.y) + w1 * bf2f(o1.y);
        out.z = w0 * bf2f(o0.z) + w1 * bf2f(o1.z);
        out.w = w0 * bf2f(o0.w) + w1 * bf2f(o1.w);
    }
    *(float4*)(y + (size_t)t * DDIM + d) = out;
}

// ---------------------------------------------------------------------------
// Host launcher
// ---------------------------------------------------------------------------
extern "C" void kernel_launch(void* const* d_in, const int* in_sizes, int n_in,
                              void* d_out, int out_size, void* d_ws, size_t ws_size,
                              hipStream_t stream) {
    const float* x  = (const float*)d_in[0];   // [BT, D]
    const float* Wg = (const float*)d_in[1];   // [D, N]
    const float* W1 = (const float*)d_in[2];   // [N, D, H]
    const float* b1 = (const float*)d_in[3];   // [N, H]
    const float* W2 = (const float*)d_in[4];   // [N, H, D]
    const float* b2 = (const float*)d_in[5];   // [N, D]
    float* y = (float*)d_out;

    // workspace carve-up (256B aligned)
    char* ws = (char*)d_ws;
    size_t off = 0;
    auto take = [&](size_t bytes) -> void* {
        void* p = ws + off;
        off = (off + bytes + 255) & ~(size_t)255;
        return p;
    };
    int*            expert_id  = (int*)           take((size_t)BTK * 4);
    float*          gate_w     = (float*)         take((size_t)BTK * 4);
    int*            dest       = (int*)           take((size_t)BTK * 4);
    int*            slot_token = (int*)           take((size_t)NC * 4);
    unsigned short* W1t        = (unsigned short*)take((size_t)NEXP * DDIM * HDIM * 2);
    unsigned short* W2t        = (unsigned short*)take((size_t)NEXP * DDIM * HDIM * 2);
    unsigned short* bufA       = (unsigned short*)take((size_t)NC * DDIM * 2); // also o-buf
    unsigned short* hbuf       = (unsigned short*)take((size_t)NC * HDIM * 2);

    // 1) weight transpose + bf16 convert (NT layout for WMMA)
    {
        dim3 g(HDIM / 32, DDIM / 32, NEXP), b(32, 8);
        hipLaunchKernelGGL(k_transpose_cvt, g, b, 0, stream, W1, W1t, DDIM, HDIM);
        hipLaunchKernelGGL(k_transpose_cvt, g, b, 0, stream, W2, W2t, HDIM, DDIM);
    }
    // 2) gating (softmax + top2 + normalize)
    hipLaunchKernelGGL(k_gate, dim3(BT / 8), dim3(256), 0, stream,
                       x, Wg, expert_id, gate_w);
    // 3) order-preserving capacity ranking
    hipLaunchKernelGGL(k_rank, dim3(NEXP), dim3(256), 0, stream,
                       expert_id, dest, slot_token);
    // 4) dispatch into bf16 capacity buffers
    hipLaunchKernelGGL(k_dispatch, dim3(NC), dim3(128), 0, stream,
                       x, slot_token, bufA);
    // 5) expert FFN layer 1: h = relu(buf @ W1 + b1)   [WMMA bf16]
    hipLaunchKernelGGL(HIP_KERNEL_NAME(k_gemm_bf16<1>),
                       dim3(CAP / 128, HDIM / 64, NEXP), dim3(256), 0, stream,
                       bufA, W1t, b1, hbuf, HDIM, CAP);
    // 6) expert FFN layer 2: o = h @ W2 + b2           [WMMA bf16, out aliases bufA]
    hipLaunchKernelGGL(HIP_KERNEL_NAME(k_gemm_bf16<0>),
                       dim3(CAP / 128, DDIM / 64, NEXP), dim3(256), 0, stream,
                       hbuf, W2t, b2, bufA, DDIM, CAP);
    // 7) weighted combine + passthrough
    hipLaunchKernelGGL(k_combine, dim3(BT), dim3(128), 0, stream,
                       x, bufA, dest, gate_w, y);
}
